// GraphAttentionLayerSparse_78769700208702
// MI455X (gfx1250) — matrix operand order repaired and checked by reading, not compiled, not run on hardware
//
#include <hip/hip_runtime.h>

#define N_NODES 50000
#define N_EDGES 500000
#define FEAT    128
#define DHEAD   64
#define HEADS   4
#define LEAKY   0.2f

typedef __attribute__((ext_vector_type(16))) __bf16 v16bf;
typedef __attribute__((ext_vector_type(2)))  __bf16 v2bf;
typedef __attribute__((ext_vector_type(8)))  float  v8f;

// ---------- order-preserving float <-> uint mapping for atomicMax ----------
__device__ __forceinline__ unsigned mapf(float f) {
    unsigned u = __float_as_uint(f);
    return (u & 0x80000000u) ? ~u : (u | 0x80000000u);
}
__device__ __forceinline__ float unmapf(unsigned u) {
    unsigned b = (u & 0x80000000u) ? (u & 0x7FFFFFFFu) : ~u;
    return __uint_as_float(b);
}

// ---------- kernel 0: zero accumulators (d_out is poisoned each run) ----------
__global__ void k_init(float* __restrict__ out, float* __restrict__ denom,
                       unsigned* __restrict__ segmax) {
    long long i = (long long)blockIdx.x * blockDim.x + threadIdx.x;
    const long long OUT = (long long)N_NODES * HEADS * DHEAD;
    if (i < OUT) out[i] = 0.0f;
    if (i < (long long)HEADS * N_NODES) { denom[i] = 0.0f; segmax[i] = 0u; }
}

// ---------- kernel 1: h[h][n][d] = X[n][:] * W[h][:][d]  (bf16 WMMA, f32 acc) ----------
// B tile kept in LDS in WMMA *fragment order*:
//   Bs[ (khi*16 + col)*16 + jj ] = W[khi*16 + jj][n0 + col],  khi = k/16
// so each lane's 16 K-values (one v16bf) are 32 contiguous, 32B-aligned bytes
// -> 2x ds_load_b128 per fragment instead of 16x ds_load_u16.
__global__ void k_gemm(const float* __restrict__ X, const float* __restrict__ W,
                       float* __restrict__ Hws) {
    __shared__ __align__(32) __bf16 Bs[FEAT * 16];   // 128x16 bf16 tile of W (4 KB)
    const int head = blockIdx.z;
    const int n0   = blockIdx.y * 16;

    // stage B tile to LDS, converting f32 -> bf16, packed in pairs (ds_store_b32)
    const float* Wh = W + (size_t)head * FEAT * DHEAD;
    for (int p = threadIdx.x; p < (FEAT * 16) / 2; p += blockDim.x) {
        int d0     = p * 2;
        int jj     = d0 & 15;            // K within 16-run (even)
        int colgrp = d0 >> 4;            // khi*16 + col
        int col    = colgrp & 15;
        int khi    = colgrp >> 4;
        int k      = khi * 16 + jj;
        v2bf pk;
        pk[0] = (__bf16)Wh[(k    ) * DHEAD + n0 + col];
        pk[1] = (__bf16)Wh[(k + 1) * DHEAD + n0 + col];
        *(v2bf*)&Bs[d0] = pk;
    }
    __syncthreads();

    const int wave   = threadIdx.x >> 5;             // 8 waves / block (wave32)
    const int lane   = threadIdx.x & 31;
    const int m_tile = blockIdx.x * 8 + wave;
    if (m_tile * 16 >= N_NODES) return;              // N_NODES % 16 == 0 -> full tiles
    const int m0     = m_tile * 16;
    const int rlo    = lane & 15;
    const int hiHalf = lane >> 4;

    v8f c = {};
    const float* ap_base = X + (size_t)(m0 + rlo) * FEAT;
    #pragma unroll
    for (int kb = 0; kb < FEAT; kb += 32) {
        v16bf a;
        // A 16x32 bf16, this kernel's contiguous K order (same as B): lanes 0-15 hold K {0..15}, lanes 16-31 hold K {16..31}
        const float* ap = ap_base + kb + hiHalf * 16;
        #pragma unroll
        for (int j = 0; j < 8; ++j) {
            a[j]     = (__bf16)ap[j];
            a[j + 8] = (__bf16)ap[j + 8];
        }
        // B fragment: one aligned 32-byte vector load from LDS
        const int khi = (kb >> 4) + hiHalf;          // covers K = kb + hiHalf*16 + [0..15]
        v16bf b = *(const v16bf*)&Bs[(khi * 16 + rlo) * 16];

        c = __builtin_amdgcn_wmma_f32_16x16x32_bf16(false, a, false, b,
                                                    (short)0, c, false, false);
    }
    // C/D layout: VGPR i -> M = i + 8*(lane>=16), col = lane%16
    float* Hh = Hws + (size_t)head * N_NODES * DHEAD;
    const int col = n0 + rlo;
    #pragma unroll
    for (int i = 0; i < 8; ++i) {
        int m = m0 + i + hiHalf * 8;
        Hh[(size_t)m * DHEAD + col] = c[i];
    }
}

// ---------- kernel 2: per-node attention half-logits ----------
__global__ void k_logits(const float* __restrict__ Hws, const float* __restrict__ A,
                         float* __restrict__ el, float* __restrict__ er) {
    int i = blockIdx.x * blockDim.x + threadIdx.x;
    if (i >= HEADS * N_NODES) return;
    int h = i / N_NODES, n = i - h * N_NODES;
    const float4* hp = (const float4*)(Hws + ((size_t)h * N_NODES + n) * DHEAD);
    const float4* al = (const float4*)(A + (size_t)h * 2 * DHEAD);
    const float4* ar = al + (DHEAD / 4);
    float sl = 0.0f, sr = 0.0f;
    #pragma unroll
    for (int d = 0; d < DHEAD / 4; ++d) {
        float4 v = hp[d], l = al[d], r = ar[d];
        sl += v.x * l.x + v.y * l.y + v.z * l.z + v.w * l.w;
        sr += v.x * r.x + v.y * r.y + v.z * r.z + v.w * r.w;
    }
    el[i] = sl; er[i] = sr;
}

// ---------- kernel 3: edge logits + segment max ----------
__global__ void k_edge_max(const int* __restrict__ ei, const float* __restrict__ el,
                           const float* __restrict__ er, float* __restrict__ ews,
                           unsigned* __restrict__ segmax) {
    int i = blockIdx.x * blockDim.x + threadIdx.x;
    if (i >= HEADS * N_EDGES) return;
    int h = i / N_EDGES, e = i - h * N_EDGES;
    int s = ei[e], d = ei[N_EDGES + e];
    float x = el[h * N_NODES + d] + er[h * N_NODES + s];
    x = (x > 0.0f) ? x : LEAKY * x;
    ews[i] = x;
    atomicMax(&segmax[h * N_NODES + d], mapf(x));
}

// ---------- kernel 4: exp + segment sum ----------
__global__ void k_edge_exp(const int* __restrict__ ei, float* __restrict__ ews,
                           const unsigned* __restrict__ segmax, float* __restrict__ denom) {
    int i = blockIdx.x * blockDim.x + threadIdx.x;
    if (i >= HEADS * N_EDGES) return;
    int h = i / N_EDGES, e = i - h * N_EDGES;
    int d = ei[N_EDGES + e];
    float ev = __expf(ews[i] - unmapf(segmax[h * N_NODES + d]));
    ews[i] = ev;
    atomicAdd(&denom[h * N_NODES + d], ev);
}

// ---------- kernel 5: alpha * edge_weight ----------
__global__ void k_edge_coef(const int* __restrict__ ei, const float* __restrict__ ew,
                            float* __restrict__ ews, const float* __restrict__ denom) {
    int i = blockIdx.x * blockDim.x + threadIdx.x;
    if (i >= HEADS * N_EDGES) return;
    int h = i / N_EDGES, e = i - h * N_EDGES;
    int d = ei[N_EDGES + e];
    ews[i] = ews[i] / (denom[h * N_NODES + d] + 1e-9f) * ew[e];
}

// ---------- kernel 6: message scatter (L2-resident gather + f32 atomics) ----------
__global__ void k_scatter(const int* __restrict__ ei, const float* __restrict__ ews,
                          const float* __restrict__ Hws, float* __restrict__ out) {
    long long t = (long long)blockIdx.x * blockDim.x + threadIdx.x;
    const long long TOT = (long long)HEADS * N_EDGES * (DHEAD / 4);
    if (t >= TOT) return;
    int       q  = (int)(t & 15);          // which float4 of the 64-wide head dim
    long long eh = t >> 4;
    int h = (int)(eh / N_EDGES), e = (int)(eh - (long long)h * N_EDGES);
    int s = ei[e], d = ei[N_EDGES + e];

    // prefetch the gather row of an edge ~2048 ahead (global_prefetch_b8)
    int ep = e + 2048;
    if (ep < N_EDGES) {
        int sp = ei[ep];
        __builtin_prefetch(Hws + ((size_t)h * N_NODES + sp) * DHEAD + q * 4, 0, 0);
    }

    float coef = ews[eh];
    const float4 v = *(const float4*)(Hws + ((size_t)h * N_NODES + s) * DHEAD + q * 4);
    float* op = out + (size_t)d * (HEADS * DHEAD) + h * DHEAD + q * 4;
    atomicAdd(op + 0, coef * v.x);
    atomicAdd(op + 1, coef * v.y);
    atomicAdd(op + 2, coef * v.z);
    atomicAdd(op + 3, coef * v.w);
}

// ---------- kernel 7: final ReLU in place ----------
__global__ void k_relu(float* __restrict__ out) {
    long long i = (long long)blockIdx.x * blockDim.x + threadIdx.x;
    const long long OUT = (long long)N_NODES * HEADS * DHEAD;
    if (i < OUT) out[i] = fmaxf(out[i], 0.0f);
}

extern "C" void kernel_launch(void* const* d_in, const int* in_sizes, int n_in,
                              void* d_out, int out_size, void* d_ws, size_t ws_size,
                              hipStream_t stream) {
    const int*   ei = (const int*)d_in[0];    // edge_index [2, E]
    const float* ew = (const float*)d_in[1];  // edge_weight [E]
    const float* X  = (const float*)d_in[2];  // features [1, N, F]
    const float* W  = (const float*)d_in[3];  // kernels [H, F, D]
    const float* A  = (const float*)d_in[4];  // attn_kernels [H, 2D, 1]
    float* out = (float*)d_out;

    // workspace layout (floats / u32, all 4B): ~62.4 MB total
    float*    Hws    = (float*)d_ws;                                   // H*N*D
    float*    el     = Hws + (size_t)HEADS * N_NODES * DHEAD;          // H*N
    float*    er     = el + (size_t)HEADS * N_NODES;                   // H*N
    unsigned* segmax = (unsigned*)(er + (size_t)HEADS * N_NODES);      // H*N
    float*    denom  = (float*)(segmax + (size_t)HEADS * N_NODES);     // H*N
    float*    ews    = denom + (size_t)HEADS * N_NODES;                // H*E

    const long long OUT = (long long)N_NODES * HEADS * DHEAD;          // 12.8M
    k_init<<<(unsigned)((OUT + 255) / 256), 256, 0, stream>>>(out, denom, segmax);

    const int m_tiles = N_NODES / 16;                                  // 3125
    dim3 gGrid((m_tiles + 7) / 8, DHEAD / 16, HEADS);                  // 391 x 4 x 4
    k_gemm<<<gGrid, 256, 0, stream>>>(X, W, Hws);

    k_logits<<<(HEADS * N_NODES + 255) / 256, 256, 0, stream>>>(Hws, A, el, er);

    const int HE = HEADS * N_EDGES;                                    // 2M
    k_edge_max <<<(HE + 255) / 256, 256, 0, stream>>>(ei, el, er, ews, segmax);
    k_edge_exp <<<(HE + 255) / 256, 256, 0, stream>>>(ei, ews, segmax, denom);
    k_edge_coef<<<(HE + 255) / 256, 256, 0, stream>>>(ei, ew, ews, denom);

    const long long SC = (long long)HE * (DHEAD / 4);                  // 32M
    k_scatter<<<(unsigned)((SC + 255) / 256), 256, 0, stream>>>(ei, ews, Hws, out);

    k_relu<<<(unsigned)((OUT + 255) / 256), 256, 0, stream>>>(out);
}
